// ObservationTransformer_89970974916875
// MI455X (gfx1250) — matrix operand order repaired and checked by reading, not compiled, not run on hardware
//
#include <hip/hip_runtime.h>
#include <hip/hip_bf16.h>
#include <cstdint>

#define DEV __device__ __forceinline__

typedef __attribute__((ext_vector_type(16))) __bf16 v16bf;
typedef __attribute__((ext_vector_type(8)))  float  v8f;

static constexpr int Bn   = 8;
static constexpr int Ln   = 1024;
static constexpr int Dm   = 1024;
static constexpr int Hn   = 16;
static constexpr int Dq   = 64;
static constexpr int Ff   = 4096;
static constexpr int NL   = 6;
static constexpr int NACT = 16;

union Frag16 { uint4 u[2]; v16bf v; };

DEV unsigned short f2bf(float f) {
  unsigned int u = __float_as_uint(f);
  u += 0x7FFFu + ((u >> 16) & 1u);          // round-to-nearest-even
  return (unsigned short)(u >> 16);
}

DEV v8f wmma_bf16(const Frag16& a, const Frag16& b, v8f c) {
  // D = A(16x32 bf16) * B(32x16 bf16) + C(16x16 f32)
  return __builtin_amdgcn_wmma_f32_16x16x32_bf16(
      false, a.v, false, b.v, (short)0, c, false, false);
}

// ---------------------------------------------------------------------------
// Embedding + positional encoding: h = embed[x]*sqrt(D) + pe ; also bf16 copy
// ---------------------------------------------------------------------------
__global__ void embed_pe_kernel(const int* __restrict__ x,
                                const float* __restrict__ embed,
                                float* __restrict__ h32,
                                unsigned short* __restrict__ hbf) {
  const int row = blockIdx.x;              // b*L + l
  const int l   = row & (Ln - 1);
  const int tok = x[row];
  const float scale = 32.0f;               // sqrt(1024)
#pragma unroll
  for (int j = 0; j < Dm / 256; ++j) {
    const int d  = threadIdx.x + j * 256;
    const int i2 = d >> 1;
    const float power   = (float)(2 * i2) / (float)Dm;
    const float divisor = __powf(10000.0f, power);
    const float ang     = (float)(l + 1) / divisor;
    const float pe      = (d & 1) ? __cosf(ang) : __sinf(ang);
    const float v = embed[(size_t)tok * Dm + d] * scale + pe;
    h32[(size_t)row * Dm + d] = v;
    hbf[(size_t)row * Dm + d] = f2bf(v);
  }
}

// ---------------------------------------------------------------------------
// mask [B,L,L] int32  ->  bitmask words (wave32 ballot: 32 cols per word)
// ---------------------------------------------------------------------------
__global__ void mask_to_bits(const int* __restrict__ mask,
                             unsigned* __restrict__ mbits) {
  const int idx = blockIdx.x * blockDim.x + threadIdx.x;  // element index
  const unsigned long long bal = __ballot(mask[idx] != 0);
  if ((threadIdx.x & 31) == 0) mbits[idx >> 5] = (unsigned)bal;
}

// ---------------------------------------------------------------------------
// Repack W[H, D, Dq] fp32  ->  WT[n=H*Dq][m=D] bf16 (head-major N)
// ---------------------------------------------------------------------------
__global__ void repack_head_wT(const float* __restrict__ W,
                               unsigned short* __restrict__ WT) {
  const int idx = blockIdx.x * blockDim.x + threadIdx.x;   // over D*H*Dq
  const int m  = idx & (Dm - 1);
  const int n  = idx / Dm;
  const int hh = n / Dq;
  const int dq = n & (Dq - 1);
  WT[(size_t)n * Dm + m] = f2bf(W[((size_t)hh * Dm + m) * Dq + dq]);
}

// ---------------------------------------------------------------------------
// Repack W[M, N] fp32 -> WT[N, M] bf16
// ---------------------------------------------------------------------------
__global__ void repack_mnT(const float* __restrict__ W,
                           unsigned short* __restrict__ WT, int M, int N) {
  const int idx = blockIdx.x * blockDim.x + threadIdx.x;   // over M*N
  const int n = idx % N;
  const int m = idx / N;
  WT[(size_t)n * M + m] = f2bf(W[(size_t)m * N + n]);
}

// ---------------------------------------------------------------------------
// vT[b][h][dq][l] = v_bf[b][l][h*Dq+dq]   (so V^T rows are contiguous in seq)
// ---------------------------------------------------------------------------
__global__ void build_vT(const unsigned short* __restrict__ vbf,
                         unsigned short* __restrict__ vT) {
  const int idx = blockIdx.x * blockDim.x + threadIdx.x;   // over B*H*Dq*L
  const int k   = idx & (Ln - 1);
  int r = idx >> 10;
  const int dq = r & (Dq - 1);  r >>= 6;
  const int hh = r & (Hn - 1);  r >>= 4;
  const int b  = r;
  vT[idx] = vbf[((size_t)(b * Ln) + k) * Dm + hh * Dq + dq];
}

// ---------------------------------------------------------------------------
// WMMA GEMM: C[M,N] = act(A[M,K] * BT[N,K]^T + bias)
// Block = 8 waves (2 M-waves x 4 N-waves), block tile 64x256, wave tile 32x64.
// A tile (64x32) double-buffered through LDS with async global->LDS copies.
// ---------------------------------------------------------------------------
template <bool RELU, bool BF16OUT>
__global__ void gemm_bf16_wmma(const unsigned short* __restrict__ A,
                               const unsigned short* __restrict__ BT,
                               const float* __restrict__ bias,
                               float* __restrict__ Cf,
                               unsigned short* __restrict__ Cbf,
                               int M, int N, int K) {
  __shared__ __align__(16) unsigned short bufA[2][64 * 32];

  const int tid   = threadIdx.x;
  const int lane  = tid & 31;
  const int wid   = tid >> 5;
  const int mw    = wid & 1;             // 2 M-wave groups
  const int nw    = wid >> 1;            // 4 N-wave groups
  const int mblk  = blockIdx.y * 64;
  const int nbase = blockIdx.x * 256 + nw * 64;
  const int hs    = lane >> 4;
  const int nloc  = lane & 15;
  const int m16   = lane & 15;

  // async A staging: thread -> (row 0..63, 16B segment 0..3)
  const int arow = tid >> 2;
  const int aseg = tid & 3;
  const unsigned short* Ag = A + (size_t)(mblk + arow) * K + aseg * 8;

  auto issueA = [&](int buf, int k0) {
    const unsigned dst =
        (unsigned)(uintptr_t)&bufA[buf][arow * 32 + aseg * 8];
    const unsigned short* src = Ag + k0;
    asm volatile("global_load_async_to_lds_b128 %0, %1, off"
                 :: "v"(dst), "v"(src) : "memory");
  };

  v8f zero = {};
  v8f acc[2][4];
#pragma unroll
  for (int i = 0; i < 2; ++i)
#pragma unroll
    for (int nt = 0; nt < 4; ++nt) acc[i][nt] = zero;

  issueA(0, 0);

  for (int k0 = 0; k0 < K; k0 += 32) {
    const int cur = (k0 >> 5) & 1;
    asm volatile("s_wait_asynccnt 0x0" ::: "memory");
    __syncthreads();                       // A tile `cur` visible to all waves
    if (k0 + 32 < K) issueA(cur ^ 1, k0 + 32);

    const unsigned short* Abuf = bufA[cur];
    Frag16 fa[2];
#pragma unroll
    for (int i = 0; i < 2; ++i) {
      const unsigned short* pp = Abuf + (mw * 32 + i * 16 + m16) * 32;
      fa[i].u[0] = *(const uint4*)(pp + hs * 8);         // a[0..7]:  k=hs*8+e
      fa[i].u[1] = *(const uint4*)(pp + 16 + hs * 8);    // a[8..15]: k=16+hs*8+e
    }
#pragma unroll
    for (int nt = 0; nt < 4; ++nt) {
      const unsigned short* Bp =
          BT + (size_t)(nbase + nt * 16 + nloc) * K + k0;
      Frag16 fb;
      fb.u[0] = *(const uint4*)(Bp + hs * 8);            // b[0..7]:  k=hs*8+e
      fb.u[1] = *(const uint4*)(Bp + 16 + hs * 8);       // b[8..15]: k=16+hs*8+e
      acc[0][nt] = wmma_bf16(fa[0], fb, acc[0][nt]);
      acc[1][nt] = wmma_bf16(fa[1], fb, acc[1][nt]);
    }
  }

#pragma unroll
  for (int i = 0; i < 2; ++i) {
#pragma unroll
    for (int nt = 0; nt < 4; ++nt) {
      const int n = nbase + nt * 16 + nloc;
      const float bv = bias ? bias[n] : 0.0f;
#pragma unroll
      for (int r = 0; r < 8; ++r) {
        const int m = mblk + mw * 32 + i * 16 + r + 8 * hs;
        float v = acc[i][nt][r] + bv;
        if (RELU) v = fmaxf(v, 0.0f);
        if (BF16OUT) Cbf[(size_t)m * N + n] = f2bf(v);
        else         Cf [(size_t)m * N + n] = v;
      }
    }
  }
}

// ---------------------------------------------------------------------------
// Flash attention: 1 wave per (b,h,16-row Q tile). 32 keys per iteration.
// ---------------------------------------------------------------------------
__global__ void attention_kernel(const unsigned short* __restrict__ qbf,
                                 const unsigned short* __restrict__ kbf,
                                 const unsigned short* __restrict__ vT,
                                 const unsigned* __restrict__ mbits,
                                 unsigned short* __restrict__ ctxbf) {
  __shared__ __align__(16) unsigned short ptile[16 * 32];

  const int lane = threadIdx.x & 31;
  const int bh   = blockIdx.x;           // b*H + h
  const int qt   = blockIdx.y;
  const int b    = bh >> 4;
  const int hh   = bh & 15;
  const int row0 = qt * 16;
  const int hs   = lane >> 4;
  const int nloc = lane & 15;
  const int m16  = lane & 15;

  // Q fragments for head-dim 0..31 and 32..63
  const unsigned short* Qrow =
      qbf + ((size_t)(b * Ln) + row0 + m16) * Dm + hh * Dq;
  Frag16 aq0, aq1;
  aq0.u[0] = *(const uint4*)(Qrow + hs * 16);
  aq0.u[1] = *(const uint4*)(Qrow + 8 + hs * 16);
  aq1.u[0] = *(const uint4*)(Qrow + 32 + hs * 16);
  aq1.u[1] = *(const uint4*)(Qrow + 40 + hs * 16);

  float mrun[8], lrun[8];
#pragma unroll
  for (int r = 0; r < 8; ++r) { mrun[r] = -1e30f; lrun[r] = 0.0f; }
  v8f zero = {};
  v8f acc[4];
#pragma unroll
  for (int i = 0; i < 4; ++i) acc[i] = zero;

  for (int c0 = 0; c0 < Ln; c0 += 32) {
    // ---- scores: two 16x16 tiles over 32 keys -----------------------------
    const unsigned short* Kp0 =
        kbf + ((size_t)(b * Ln) + c0 + nloc) * Dm + hh * Dq;
    const unsigned short* Kp1 = Kp0 + (size_t)16 * Dm;
    Frag16 fb;
    v8f s0 = zero, s1 = zero;
    fb.u[0] = ((const uint4*)(Kp0 + hs * 16))[0];
    fb.u[1] = ((const uint4*)(Kp0 + hs * 16))[1];
    s0 = wmma_bf16(aq0, fb, s0);
    fb.u[0] = ((const uint4*)(Kp0 + 32 + hs * 16))[0];
    fb.u[1] = ((const uint4*)(Kp0 + 32 + hs * 16))[1];
    s0 = wmma_bf16(aq1, fb, s0);
    fb.u[0] = ((const uint4*)(Kp1 + hs * 16))[0];
    fb.u[1] = ((const uint4*)(Kp1 + hs * 16))[1];
    s1 = wmma_bf16(aq0, fb, s1);
    fb.u[0] = ((const uint4*)(Kp1 + 32 + hs * 16))[0];
    fb.u[1] = ((const uint4*)(Kp1 + 32 + hs * 16))[1];
    s1 = wmma_bf16(aq1, fb, s1);

    // ---- scale + bitmask + online softmax ---------------------------------
#pragma unroll
    for (int r = 0; r < 8; ++r) {
      const int mrowg = row0 + r + 8 * hs;
      const unsigned mb = mbits[((size_t)b * Ln + mrowg) * 32 + (c0 >> 5)];
      float v0 = s0[r] * 0.125f;
      float v1 = s1[r] * 0.125f;
      if (!((mb >> nloc) & 1u))        v0 = -1e9f;
      if (!((mb >> (16 + nloc)) & 1u)) v1 = -1e9f;

      float mx = fmaxf(v0, v1);
#pragma unroll
      for (int off = 8; off >= 1; off >>= 1)
        mx = fmaxf(mx, __shfl_xor(mx, off, 32));
      const float mnew = fmaxf(mrun[r], mx);
      const float p0 = __expf(v0 - mnew);
      const float p1 = __expf(v1 - mnew);
      float ls = p0 + p1;
#pragma unroll
      for (int off = 8; off >= 1; off >>= 1)
        ls += __shfl_xor(ls, off, 32);
      const float corr = __expf(mrun[r] - mnew);
      lrun[r] = lrun[r] * corr + ls;
      mrun[r] = mnew;
      acc[0][r] *= corr; acc[1][r] *= corr;
      acc[2][r] *= corr; acc[3][r] *= corr;

      const int lm = r + 8 * hs;                 // C-layout row -> LDS row
      ptile[lm * 32 + nloc]      = f2bf(p0);
      ptile[lm * 32 + 16 + nloc] = f2bf(p1);
    }
    __syncthreads();

    // ---- reload probs as A fragment (16x32) -------------------------------
    Frag16 ap;
    const unsigned short* pp = ptile + m16 * 32;
    ap.u[0] = *(const uint4*)(pp + hs * 16);
    ap.u[1] = *(const uint4*)(pp + 8 + hs * 16);

    // ---- acc += P(16x32) * V(32x16 per dim-tile) --------------------------
#pragma unroll
    for (int nt = 0; nt < 4; ++nt) {
      const unsigned short* Vp =
          vT + ((size_t)(b * Hn + hh) * Dq + nt * 16 + nloc) * Ln + c0 + hs * 16;
      Frag16 bv;
      bv.u[0] = ((const uint4*)Vp)[0];
      bv.u[1] = ((const uint4*)Vp)[1];
      acc[nt] = wmma_bf16(ap, bv, acc[nt]);
    }
    __syncthreads();
  }

  // ---- normalize + store ctx (bf16, head-major) ---------------------------
#pragma unroll
  for (int nt = 0; nt < 4; ++nt) {
#pragma unroll
    for (int r = 0; r < 8; ++r) {
      const int m = row0 + r + 8 * hs;
      const float v = acc[nt][r] / lrun[r];
      ctxbf[((size_t)(b * Ln) + m) * Dm + hh * Dq + nt * 16 + nloc] = f2bf(v);
    }
  }
}

// ---------------------------------------------------------------------------
// h = LayerNorm(h + res) * s + b ; also refresh bf16 mirror
// ---------------------------------------------------------------------------
__global__ void add_ln_kernel(float* __restrict__ h32,
                              const float* __restrict__ res,
                              const float* __restrict__ sc,
                              const float* __restrict__ bi,
                              unsigned short* __restrict__ hbf) {
  __shared__ float sred[256];
  const int row = blockIdx.x;
  const int tid = threadIdx.x;
  float v[Dm / 256];
  float sum = 0.0f;
#pragma unroll
  for (int j = 0; j < Dm / 256; ++j) {
    const int d = tid + j * 256;
    v[j] = h32[(size_t)row * Dm + d] + res[(size_t)row * Dm + d];
    sum += v[j];
  }
  sred[tid] = sum; __syncthreads();
  for (int st = 128; st > 0; st >>= 1) {
    if (tid < st) sred[tid] += sred[tid + st];
    __syncthreads();
  }
  const float mu = sred[0] / (float)Dm;
  __syncthreads();
  float sq = 0.0f;
#pragma unroll
  for (int j = 0; j < Dm / 256; ++j) { const float d0 = v[j] - mu; sq += d0 * d0; }
  sred[tid] = sq; __syncthreads();
  for (int st = 128; st > 0; st >>= 1) {
    if (tid < st) sred[tid] += sred[tid + st];
    __syncthreads();
  }
  const float rs = rsqrtf(sred[0] / (float)Dm + 1e-5f);
#pragma unroll
  for (int j = 0; j < Dm / 256; ++j) {
    const int d = tid + j * 256;
    const float o = (v[j] - mu) * rs * sc[d] + bi[d];
    h32[(size_t)row * Dm + d] = o;
    hbf[(size_t)row * Dm + d] = f2bf(o);
  }
}

// ---------------------------------------------------------------------------
// logits = log_softmax(h[:, -1, :] @ out_w + out_b)  — tiny epilogue
// ---------------------------------------------------------------------------
__global__ void logits_kernel(const float* __restrict__ h32,
                              const float* __restrict__ out_w,
                              const float* __restrict__ out_b,
                              float* __restrict__ out) {
  __shared__ float sred[256];
  __shared__ float lg[NACT];
  const int b = blockIdx.x;
  const float* hrow = h32 + ((size_t)b * Ln + (Ln - 1)) * Dm;
  for (int a = 0; a < NACT; ++a) {
    float s = 0.0f;
    for (int d = threadIdx.x; d < Dm; d += 256)
      s += hrow[d] * out_w[(size_t)d * NACT + a];
    sred[threadIdx.x] = s; __syncthreads();
    for (int st = 128; st > 0; st >>= 1) {
      if (threadIdx.x < st) sred[threadIdx.x] += sred[threadIdx.x + st];
      __syncthreads();
    }
    if (threadIdx.x == 0) lg[a] = sred[0] + out_b[a];
    __syncthreads();
  }
  if (threadIdx.x == 0) {
    float mx = -1e30f;
    for (int a = 0; a < NACT; ++a) mx = fmaxf(mx, lg[a]);
    float ss = 0.0f;
    for (int a = 0; a < NACT; ++a) ss += __expf(lg[a] - mx);
    const float lse = mx + __logf(ss);
    for (int a = 0; a < NACT; ++a) out[b * NACT + a] = lg[a] - lse;
  }
}

// ---------------------------------------------------------------------------
extern "C" void kernel_launch(void* const* d_in, const int* in_sizes, int n_in,
                              void* d_out, int out_size, void* d_ws, size_t ws_size,
                              hipStream_t stream) {
  (void)in_sizes; (void)n_in; (void)out_size; (void)ws_size;
  const int*   x     = (const int*)  d_in[0];
  const int*   mask  = (const int*)  d_in[1];
  const float* embed = (const float*)d_in[2];
  const float* Wq    = (const float*)d_in[3];
  const float* Wk    = (const float*)d_in[4];
  const float* Wv    = (const float*)d_in[5];
  const float* Wo_w  = (const float*)d_in[6];
  const float* Wo_b  = (const float*)d_in[7];
  const float* ln1_s = (const float*)d_in[8];
  const float* ln1_b = (const float*)d_in[9];
  const float* ff_w1 = (const float*)d_in[10];
  const float* ff_b1 = (const float*)d_in[11];
  const float* ff_w2 = (const float*)d_in[12];
  const float* ff_b2 = (const float*)d_in[13];
  const float* ln2_s = (const float*)d_in[14];
  const float* ln2_b = (const float*)d_in[15];
  const float* out_w = (const float*)d_in[16];
  const float* out_b = (const float*)d_in[17];

  const size_t TOK = (size_t)Bn * Ln;                 // 8192 rows
  char* ws = (char*)d_ws;
  size_t off = 0;
  auto alloc = [&](size_t bytes) { void* p = ws + off; off += bytes; return p; };

  float*          h32   = (float*)         alloc(TOK * Dm * 4);   // 32 MB
  unsigned short* hbf   = (unsigned short*)alloc(TOK * Dm * 2);   // 16 MB
  unsigned short* qbf   = (unsigned short*)alloc(TOK * Dm * 2);
  unsigned short* kbf   = (unsigned short*)alloc(TOK * Dm * 2);
  unsigned short* vbf   = (unsigned short*)alloc(TOK * Dm * 2);
  unsigned short* vTb   = (unsigned short*)alloc(TOK * Dm * 2);
  unsigned short* ctxbf = (unsigned short*)alloc(TOK * Dm * 2);
  float*          tmp32 = (float*)         alloc(TOK * Dm * 4);   // 32 MB
  unsigned short* ff1bf = (unsigned short*)alloc(TOK * Ff * 2);   // 64 MB
  unsigned short* wTa   = (unsigned short*)alloc((size_t)Ff * Dm * 2); // 8 MB
  unsigned short* wTb2  = (unsigned short*)alloc((size_t)Ff * Dm * 2); // 8 MB
  unsigned*       mbits = (unsigned*)      alloc((size_t)Bn * Ln * Ln / 8); // 1 MB
  unsigned short* wqT = wTa;                       // 2 MB each, share region
  unsigned short* wkT = wTa + (size_t)Dm * Dm;
  unsigned short* wvT = wTa + (size_t)2 * Dm * Dm;

  embed_pe_kernel<<<(int)TOK, 256, 0, stream>>>(x, embed, h32, hbf);
  mask_to_bits<<<(int)((size_t)Bn * Ln * Ln / 256), 256, 0, stream>>>(mask, mbits);

  const dim3 gD (Dm / 256, (int)TOK / 64);   // GEMM N=1024
  const dim3 gFF(Ff / 256, (int)TOK / 64);   // GEMM N=4096

  for (int i = 0; i < NL; ++i) {
    const size_t wqkv_off = (size_t)i * Hn * Dm * Dq;
    repack_head_wT<<<(Dm * Hn * Dq) / 256, 256, 0, stream>>>(Wq + wqkv_off, wqT);
    repack_head_wT<<<(Dm * Hn * Dq) / 256, 256, 0, stream>>>(Wk + wqkv_off, wkT);
    repack_head_wT<<<(Dm * Hn * Dq) / 256, 256, 0, stream>>>(Wv + wqkv_off, wvT);

    gemm_bf16_wmma<false, true><<<gD, 256, 0, stream>>>(hbf, wqT, nullptr, nullptr, qbf, (int)TOK, Dm, Dm);
    gemm_bf16_wmma<false, true><<<gD, 256, 0, stream>>>(hbf, wkT, nullptr, nullptr, kbf, (int)TOK, Dm, Dm);
    gemm_bf16_wmma<false, true><<<gD, 256, 0, stream>>>(hbf, wvT, nullptr, nullptr, vbf, (int)TOK, Dm, Dm);

    build_vT<<<(int)(TOK * Dm / 256), 256, 0, stream>>>(vbf, vTb);

    attention_kernel<<<dim3(Bn * Hn, Ln / 16), 32, 0, stream>>>(qbf, kbf, vTb, mbits, ctxbf);

    repack_mnT<<<(Dm * Dm) / 256, 256, 0, stream>>>(Wo_w + (size_t)i * Dm * Dm, wTa, Dm, Dm);
    gemm_bf16_wmma<false, false><<<gD, 256, 0, stream>>>(ctxbf, wTa, Wo_b + (size_t)i * Dm, tmp32, nullptr, (int)TOK, Dm, Dm);

    add_ln_kernel<<<(int)TOK, 256, 0, stream>>>(h32, tmp32, ln1_s + (size_t)i * Dm, ln1_b + (size_t)i * Dm, hbf);

    repack_mnT<<<(Dm * Ff) / 256, 256, 0, stream>>>(ff_w1 + (size_t)i * Dm * Ff, wTa, Dm, Ff);
    gemm_bf16_wmma<true, true><<<gFF, 256, 0, stream>>>(hbf, wTa, ff_b1 + (size_t)i * Ff, nullptr, ff1bf, (int)TOK, Ff, Dm);

    repack_mnT<<<(Ff * Dm) / 256, 256, 0, stream>>>(ff_w2 + (size_t)i * Ff * Dm, wTb2, Ff, Dm);
    gemm_bf16_wmma<false, false><<<gD, 256, 0, stream>>>(ff1bf, wTb2, ff_b2 + (size_t)i * Dm, tmp32, nullptr, (int)TOK, Dm, Ff);

    add_ln_kernel<<<(int)TOK, 256, 0, stream>>>(h32, tmp32, ln2_s + (size_t)i * Dm, ln2_b + (size_t)i * Dm, hbf);
  }

  logits_kernel<<<Bn, 256, 0, stream>>>(h32, out_w, out_b, (float*)d_out);
}
